// MultiheadAttention_63161789055037
// MI455X (gfx1250) — compile-verified, matrix operand order checked by reading, not run on hardware
//
#include <hip/hip_runtime.h>
#include <hip/hip_bf16.h>

// CDNA5 / gfx1250 attention block:
//   qh = (q·Wq^T)*1/8 ; kh = k·Wk^T + bk ; vh = v·Wv^T + bv      (f16 in ws)
//   S  = qh·kh^T (per batch) -> row softmax (f32 math, f16 store)
//   A  = P·vh (per batch)                                         (f16 in ws)
//   x  = A·Wo^T + bo + q (f32 -> d_out) ; LayerNorm in-place.
// Workspace (halves): qh, kh, vh, attn (16384*1024 each) + scores (8*2048^2)
// => ~192 MiB.

typedef __attribute__((ext_vector_type(16))) _Float16 v16h;
typedef __attribute__((ext_vector_type(8)))  _Float16 v8h;
typedef __attribute__((ext_vector_type(8)))  float    v8f;
typedef __attribute__((ext_vector_type(4)))  int      v4i;

#define BM 128
#define BN 128
#define BK 32
#define LDT 40   // padded LDS row stride (halves): 80B

// ---- gfx1250 async global->LDS copy (ASYNCcnt) with safe fallback ----------
#if defined(__HIP_DEVICE_COMPILE__) && __has_builtin(__builtin_amdgcn_global_load_async_to_lds_b128)
#define HAVE_ASYNC_LDS 1
#else
#define HAVE_ASYNC_LDS 0
#endif

#if HAVE_ASYNC_LDS
typedef __attribute__((address_space(1))) v4i* gptr_v4i;
typedef __attribute__((address_space(3))) v4i* lptr_v4i;
#endif

__device__ __forceinline__ void copy32B_g2l(_Float16* dst, const _Float16* src) {
#if HAVE_ASYNC_LDS
    __builtin_amdgcn_global_load_async_to_lds_b128(
        (gptr_v4i)(src), (lptr_v4i)(dst), 0, 0);
    __builtin_amdgcn_global_load_async_to_lds_b128(
        (gptr_v4i)(src + 8), (lptr_v4i)(dst + 8), 0, 0);
#else
    v8h a = *(const v8h*)src;
    v8h b = *(const v8h*)(src + 8);
    *(v8h*)dst       = a;
    *(v8h*)(dst + 8) = b;
#endif
}

__device__ __forceinline__ void wait_async0() {
#if HAVE_ASYNC_LDS
#if __has_builtin(__builtin_amdgcn_s_wait_asynccnt)
    __builtin_amdgcn_s_wait_asynccnt(0);
#else
    asm volatile("s_wait_asynccnt 0x0" ::: "memory");
#endif
#endif
}

// Build a 16x32 (A) / 32x16 (B) f16 WMMA fragment from an LDS tile [row][k].
// CDNA5 ISA 7.12.2: lanes 0-15: row=lane, halves 0-7 -> K0-7, 8-15 -> K16-23;
// lanes 16-31: row=lane-16, halves 0-7 -> K8-15, 8-15 -> K24-31.
__device__ __forceinline__ v16h load_frag(const _Float16 (*T)[LDT], int row, int kOff) {
    v8h lo = *(const v8h*)&T[row][kOff];
    v8h hi = *(const v8h*)&T[row][kOff + 16];
    v16h r;
#pragma unroll
    for (int i = 0; i < 8; ++i) { r[i] = lo[i]; r[i + 8] = hi[i]; }
    return r;
}

template<int N, int K, bool A_HALF, bool B_HALF, bool B_TRANS>
__device__ __forceinline__ void issue_tiles(_Float16 (*As)[LDT], _Float16 (*Bs)[LDT],
                                            const void* Ap, const void* Bp,
                                            int m0, int n0, int k0, int tid) {
    // ---- A tile: BM x BK, source row-major [m][K]
    {
        const int r  = tid >> 1;
        const int c0 = (tid & 1) * 16;
        if constexpr (A_HALF) {
            const _Float16* src = (const _Float16*)Ap + (size_t)(m0 + r) * K + k0 + c0;
            copy32B_g2l(&As[r][c0], src);
        } else {
            const float* src = (const float*)Ap + (size_t)(m0 + r) * K + k0 + c0;
#pragma unroll
            for (int i = 0; i < 16; ++i) As[r][c0 + i] = (_Float16)src[i];
        }
    }
    // ---- B tile -> Bs[n][k]
    if constexpr (!B_TRANS) {
        const int r  = tid >> 1;
        const int c0 = (tid & 1) * 16;
        if constexpr (B_HALF) {
            const _Float16* src = (const _Float16*)Bp + (size_t)(n0 + r) * K + k0 + c0;
            copy32B_g2l(&Bs[r][c0], src);
        } else {
            const float* src = (const float*)Bp + (size_t)(n0 + r) * K + k0 + c0;
#pragma unroll
            for (int i = 0; i < 16; ++i) Bs[r][c0 + i] = (_Float16)src[i];
        }
    } else {
        // B stored [K][N] row-major f16: transpose during LDS store
        const int kr = tid >> 3;           // 0..31
        const int c0 = (tid & 7) * 16;     // 0..112
        const _Float16* src = (const _Float16*)Bp + (size_t)(k0 + kr) * N + n0 + c0;
#pragma unroll
        for (int i = 0; i < 16; ++i) Bs[c0 + i][kr] = src[i];
    }
}

template<int N, int K, bool A_HALF, bool B_HALF, bool OUT_HALF, bool B_TRANS,
         bool HAS_BIAS, bool HAS_RESID, bool SCALE>
__global__ __launch_bounds__(256)
void gemm_nt(const void* __restrict__ Ap, const void* __restrict__ Bp,
             void* __restrict__ Cp,
             long strideA, long strideB, long strideC,   // per-batch strides (elements)
             float alpha,
             const float* __restrict__ bias,             // [N]
             const float* __restrict__ resid)            // [M,N] f32
{
    __shared__ __align__(16) _Float16 As[2][BM][LDT];
    __shared__ __align__(16) _Float16 Bs[2][BN][LDT];

    const int tid  = threadIdx.x;
    const int lane = tid & 31;                 // wave32
    const int wave = tid >> 5;                 // 8 waves
    const int z    = blockIdx.z;
    const int m0   = blockIdx.y * BM;
    const int n0   = blockIdx.x * BN;

    const int mBase = (wave >> 1) * 32;        // 4 wave-rows x 2 wave-cols
    const int nBase = (wave & 1) * 64;
    const int l15   = lane & 15;
    const int kOff  = (lane & 16) ? 8 : 0;
    const int mOff  = (lane & 16) ? 8 : 0;

    const void* Az = (const void*)((const char*)Ap + (size_t)z * strideA * (A_HALF ? 2 : 4));
    const void* Bz = (const void*)((const char*)Bp + (size_t)z * strideB * (B_HALF ? 2 : 4));

    constexpr bool USES_ASYNC = (A_HALF || (B_HALF && !B_TRANS));
    constexpr int  NT = K / BK;

    v8f acc[2][4] = {};

    issue_tiles<N, K, A_HALF, B_HALF, B_TRANS>(As[0], Bs[0], Az, Bz, m0, n0, 0, tid);

    for (int kt = 0; kt < NT; ++kt) {
        const int cur = kt & 1;
        if constexpr (USES_ASYNC) wait_async0();
        __syncthreads();
        if (kt + 1 < NT)
            issue_tiles<N, K, A_HALF, B_HALF, B_TRANS>(As[cur ^ 1], Bs[cur ^ 1],
                                                       Az, Bz, m0, n0, (kt + 1) * BK, tid);
        v16h a[2], b[4];
#pragma unroll
        for (int mi = 0; mi < 2; ++mi) a[mi] = load_frag(As[cur], mBase + mi * 16 + l15, kOff);
#pragma unroll
        for (int ni = 0; ni < 4; ++ni) b[ni] = load_frag(Bs[cur], nBase + ni * 16 + l15, kOff);
#pragma unroll
        for (int mi = 0; mi < 2; ++mi)
#pragma unroll
            for (int ni = 0; ni < 4; ++ni)
                acc[mi][ni] = __builtin_amdgcn_wmma_f32_16x16x32_f16(
                    false, a[mi], false, b[ni], (short)0, acc[mi][ni], false, false);
    }

    // ---- epilogue (C/D layout: VGPR r -> M = mOff + r, lane -> N)
#pragma unroll
    for (int mi = 0; mi < 2; ++mi) {
        const int mrow = m0 + mBase + mi * 16 + mOff;
#pragma unroll
        for (int ni = 0; ni < 4; ++ni) {
            const int n = n0 + nBase + ni * 16 + l15;
            const float bb = HAS_BIAS ? bias[n] : 0.0f;
            const float* rp = HAS_RESID ? (resid + (size_t)mrow * N + n) : nullptr;
#pragma unroll
            for (int r = 0; r < 8; ++r) {
                float vv = acc[mi][ni][r];
                if constexpr (SCALE)     vv *= alpha;
                if constexpr (HAS_BIAS)  vv += bb;
                if constexpr (HAS_RESID) vv += rp[r * N];
                if constexpr (OUT_HALF) {
                    _Float16* cp = (_Float16*)Cp + (size_t)z * strideC + (size_t)mrow * N + n;
                    cp[r * N] = (_Float16)vv;
                } else {
                    float* cp = (float*)Cp + (size_t)z * strideC + (size_t)mrow * N + n;
                    cp[r * N] = vv;
                }
            }
        }
    }
}

__global__ __launch_bounds__(256)
void softmax_rows(_Float16* __restrict__ s, int cols) {
    __shared__ float red[256];
    _Float16* p = s + (size_t)blockIdx.x * cols;
    const int tid = threadIdx.x;

    float lmax = -3.0e38f;
    for (int c = tid; c < cols; c += 256) lmax = fmaxf(lmax, (float)p[c]);
    red[tid] = lmax; __syncthreads();
    for (int k = 128; k > 0; k >>= 1) { if (tid < k) red[tid] = fmaxf(red[tid], red[tid + k]); __syncthreads(); }
    const float m = red[0]; __syncthreads();

    float lsum = 0.0f;
    for (int c = tid; c < cols; c += 256) lsum += __expf((float)p[c] - m);
    red[tid] = lsum; __syncthreads();
    for (int k = 128; k > 0; k >>= 1) { if (tid < k) red[tid] += red[tid + k]; __syncthreads(); }
    const float inv = 1.0f / red[0];

    for (int c = tid; c < cols; c += 256) p[c] = (_Float16)(__expf((float)p[c] - m) * inv);
}

__global__ __launch_bounds__(256)
void layernorm_inplace(float* __restrict__ x,
                       const float* __restrict__ gamma,
                       const float* __restrict__ beta, int cols) {
    __shared__ float r1[256], r2[256];
    float* p = x + (size_t)blockIdx.x * cols;
    const int tid = threadIdx.x;
    float v[4], s = 0.0f, s2 = 0.0f;
#pragma unroll
    for (int i = 0; i < 4; ++i) {
        v[i] = p[tid + i * 256];
        s += v[i]; s2 += v[i] * v[i];
    }
    r1[tid] = s; r2[tid] = s2; __syncthreads();
    for (int k = 128; k > 0; k >>= 1) {
        if (tid < k) { r1[tid] += r1[tid + k]; r2[tid] += r2[tid + k]; }
        __syncthreads();
    }
    const float mu  = r1[0] / cols;
    const float var = r2[0] / cols - mu * mu;
    const float inv = rsqrtf(var + 1e-5f);
#pragma unroll
    for (int i = 0; i < 4; ++i) {
        const int c = tid + i * 256;
        p[c] = (v[i] - mu) * inv * gamma[c] + beta[c];
    }
}

extern "C" void kernel_launch(void* const* d_in, const int* in_sizes, int n_in,
                              void* d_out, int out_size, void* d_ws, size_t ws_size,
                              hipStream_t stream) {
    const float* q     = (const float*)d_in[0];
    const float* k     = (const float*)d_in[1];
    const float* v     = (const float*)d_in[2];
    const float* Wq    = (const float*)d_in[3];
    const float* Wk    = (const float*)d_in[4];
    const float* bk    = (const float*)d_in[5];
    const float* Wv    = (const float*)d_in[6];
    const float* bv    = (const float*)d_in[7];
    const float* Wo    = (const float*)d_in[8];
    const float* bo    = (const float*)d_in[9];
    const float* gamma = (const float*)d_in[10];
    const float* beta  = (const float*)d_in[11];
    float* out = (float*)d_out;

    const int B = 8, S = 2048;
    constexpr int D = 1024, HD = 1024, SS = 2048;
    const int MS = B * S;                       // 16384

    _Float16* qh     = (_Float16*)d_ws;
    _Float16* kh     = qh   + (size_t)MS * HD;
    _Float16* vh     = kh   + (size_t)MS * HD;
    _Float16* attn   = vh   + (size_t)MS * HD;
    _Float16* scores = attn + (size_t)MS * HD;  // 8 * 2048 * 2048 halves

    dim3 blk(256);

    // 1) projections: [MS,D] x [HD,D]^T -> f16
    dim3 gproj(HD / BN, MS / BM, 1);
    //           N   K    Ah     Bh     OUTh  Bt     bias   resid  scale
    gemm_nt<HD, D, false, false, true, false, false, false, true><<<gproj, blk, 0, stream>>>(
        q, Wq, qh, 0, 0, 0, 0.125f, nullptr, nullptr);          // *1/sqrt(64)
    gemm_nt<HD, D, false, false, true, false, true, false, false><<<gproj, blk, 0, stream>>>(
        k, Wk, kh, 0, 0, 0, 1.0f, bk, nullptr);
    gemm_nt<HD, D, false, false, true, false, true, false, false><<<gproj, blk, 0, stream>>>(
        v, Wv, vh, 0, 0, 0, 1.0f, bv, nullptr);

    // 2) scores = qh · kh^T per batch: [S,HD] x [S,HD]^T
    dim3 gsc(SS / BN, S / BM, B);
    gemm_nt<SS, HD, true, true, true, false, false, false, false><<<gsc, blk, 0, stream>>>(
        qh, kh, scores, (long)S * HD, (long)S * HD, (long)S * SS, 1.0f, nullptr, nullptr);

    // 3) softmax over each of the 8*2048 rows (2048 wide)
    softmax_rows<<<dim3(B * S), blk, 0, stream>>>(scores, SS);

    // 4) attn = P · vh per batch: [S,SS] x [SS,HD] (B stored [K][N] -> transpose path)
    dim3 gav(HD / BN, S / BM, B);
    gemm_nt<HD, SS, true, true, true, true, false, false, false><<<gav, blk, 0, stream>>>(
        scores, vh, attn, (long)S * SS, (long)S * HD, (long)S * HD, 1.0f, nullptr, nullptr);

    // 5) out = attn · Wo^T + bo + q  (f32)
    dim3 go(D / BN, MS / BM, 1);
    gemm_nt<D, HD, true, false, false, false, true, true, false><<<go, blk, 0, stream>>>(
        attn, Wo, out, 0, 0, 0, 1.0f, bo, q);

    // 6) LayerNorm in-place on d_out rows
    layernorm_inplace<<<dim3(MS), blk, 0, stream>>>(out, gamma, beta, D);
}
